// SimpleALIFRNNTbptt_30571577213352
// MI455X (gfx1250) — matrix operand order repaired and checked by reading, not compiled, not run on hardware
//
#include <hip/hip_runtime.h>
#include <hip/hip_bf16.h>

// ---------------------------------------------------------------------------
// ALIF spiking RNN forward on gfx1250 (MI455X), WMMA f16 + TDM async path.
//
//   0) convert x -> f16 (vectorized, one pass), W_hidden -> f16 (split)
//   1) Xcur = x_h @ Wx : WMMA GEMM, A tiles DMA'd to LDS by the Tensor Data
//      Mover (tensor_load_to_lds, double buffered, TENSORcnt synchronized)
//   2) per step t: rec = z @ Wh WMMA GEMM fused with ALIF update; then
//      readout + NLL + spike count
// ---------------------------------------------------------------------------

typedef __attribute__((ext_vector_type(16))) _Float16 v16h;
typedef __attribute__((ext_vector_type(8)))  _Float16 v8h;
typedef __attribute__((ext_vector_type(8)))  float    v8f;
typedef __attribute__((ext_vector_type(4)))  unsigned int u32x4;
typedef __attribute__((ext_vector_type(8)))  int      i32x8;
typedef __attribute__((ext_vector_type(4)))  int      i32x4;

union Frag16 { v16h v; v8h h[2]; };

#define T_STEPS 250
#define BATCH   256
#define INSZ    700
#define KIN     704          // input K padded to multiple of 32
#define H       512
#define OUTSZ   10
#define MB_X    64000        // T_STEPS * BATCH

#define BM 64                // block tile M
#define BN 128               // block tile N
#define BK 32                // K chunk (== WMMA K)
#define LDA 40               // LDS row stride in halves (TDM pad: 64B + 16B)

// ---------------------------------------------------------------------------
// init: decay constants, zero states + loss/spike accumulators
// ---------------------------------------------------------------------------
__global__ __launch_bounds__(256) void init_kernel(
    const float* __restrict__ tau_m, const float* __restrict__ tau_a,
    const float* __restrict__ tau_o,
    float* __restrict__ alpha, float* __restrict__ rho, float* __restrict__ kappa,
    _Float16* __restrict__ z0, float* __restrict__ u, float* __restrict__ a,
    float* __restrict__ ou, float* __restrict__ out_tail)
{
    int idx = blockIdx.x * blockDim.x + threadIdx.x;
    int nth = gridDim.x * blockDim.x;
    if (idx < H)     { alpha[idx] = expf(-1.0f / tau_m[idx]);
                       rho[idx]   = expf(-1.0f / tau_a[idx]); }
    if (idx < OUTSZ) { kappa[idx] = expf(-1.0f / tau_o[idx]); }
    if (idx < 2)     { out_tail[idx] = 0.0f; }            // loss, spikes
    for (int i = idx; i < BATCH * H; i += nth) {
        z0[i] = (_Float16)0.0f; u[i] = 0.0f; a[i] = 0.0f;
    }
    for (int i = idx; i < BATCH * OUTSZ; i += nth) ou[i] = 0.0f;
}

// ---------------------------------------------------------------------------
// convert x (f32, [64000,700]) -> f16 [64000,704] zero-padded, vectorized:
// each thread produces 8 contiguous halves (two float4 loads, one 16B store).
// ---------------------------------------------------------------------------
__global__ __launch_bounds__(256) void convert_x_kernel(
    const float* __restrict__ x, _Float16* __restrict__ xh)
{
    const int chunk = blockIdx.x * blockDim.x + threadIdx.x;   // 64000*88 chunks
    if (chunk >= MB_X * (KIN / 8)) return;
    const int r  = chunk / (KIN / 8);
    const int c8 = (chunk % (KIN / 8)) * 8;
    v8h o;
    if (c8 + 8 <= INSZ) {
        const float4 f0 = *(const float4*)&x[(size_t)r * INSZ + c8];
        const float4 f1 = *(const float4*)&x[(size_t)r * INSZ + c8 + 4];
        o[0] = (_Float16)f0.x; o[1] = (_Float16)f0.y;
        o[2] = (_Float16)f0.z; o[3] = (_Float16)f0.w;
        o[4] = (_Float16)f1.x; o[5] = (_Float16)f1.y;
        o[6] = (_Float16)f1.z; o[7] = (_Float16)f1.w;
    } else {
        #pragma unroll
        for (int e = 0; e < 8; ++e) {
            int k = c8 + e;
            o[e] = (k < INSZ) ? (_Float16)x[(size_t)r * INSZ + k] : (_Float16)0.0f;
        }
    }
    *(v8h*)&xh[(size_t)r * KIN + c8] = o;
}

// ---------------------------------------------------------------------------
// convert W_hidden (f32, [1212,512]) -> f16 Wx [704,512] (zero pad) + Wh [512,512]
// ---------------------------------------------------------------------------
__global__ __launch_bounds__(256) void convert_w_kernel(
    const float* __restrict__ W, _Float16* __restrict__ Wx, _Float16* __restrict__ Wh)
{
    int idx = blockIdx.x * blockDim.x + threadIdx.x;
    int nth = gridDim.x * blockDim.x;
    for (int i = idx; i < KIN * H; i += nth) {
        int k = i >> 9, n = i & (H - 1);
        Wx[i] = (k < INSZ) ? (_Float16)W[(size_t)k * H + n] : (_Float16)0.0f;
    }
    for (int i = idx; i < H * H; i += nth) {
        int k = i >> 9, n = i & (H - 1);
        Wh[i] = (_Float16)W[(size_t)(INSZ + k) * H + n];
    }
}

// ---------------------------------------------------------------------------
// TDM: DMA a 64-row x 32-col f16 tile (row stride KIN halves) from global to
// LDS at lds_off, with LDS padding 16dw interval / 4dw amount -> LDA=40 halves.
// D# layout per CDNA5 ISA 8.3/8.4 (2D tensor, groups 2/3 zero).
// clang-23 toolchain: 6-arg builtin (g0, g1, g2, g3, g4, cpol).
// ---------------------------------------------------------------------------
__device__ __forceinline__ void tdm_load_tile(const _Float16* gsrc, unsigned lds_off)
{
    const unsigned long long ga = (unsigned long long)(uintptr_t)gsrc;
    u32x4 g0;
    g0[0] = 1u;                                   // count=1 valid user descriptor
    g0[1] = lds_off;                              // lds_addr (bytes)
    g0[2] = (unsigned)(ga & 0xFFFFFFFFu);         // global_addr[31:0]
    g0[3] = (unsigned)((ga >> 32) & 0x01FFFFFFu)  // global_addr[56:32]
          | (2u << 30);                           // type = 2 ("image")
    i32x8 g1;
    g1[0] = (1 << 16)                             // data_size = 2 bytes
          | (1 << 20)                             // pad_enable
          | (3 << 22)                             // pad_interval: 16 DWORDs (one row)
          | (3 << 25);                            // pad_amount: 4 DWORDs (16B)
    g1[1] = (int)((KIN & 0xFFFF) << 16);          // tensor_dim0[15:0]
    g1[2] = (int)((KIN >> 16) | ((MB_X & 0xFFFF) << 16));  // dim0 hi | dim1 lo
    g1[3] = (int)((MB_X >> 16) | (BK << 16));     // dim1 hi | tile_dim0 = 32
    g1[4] = BM;                                   // tile_dim1 = 64 rows (tile_dim2=0)
    g1[5] = (int)KIN;                             // tensor_dim0_stride[31:0]
    g1[6] = 0;                                    // stride0 hi | stride1 lo
    g1[7] = 0;
    const i32x4 z4 = {0, 0, 0, 0};
    const i32x8 z8 = {0, 0, 0, 0, 0, 0, 0, 0};
    __builtin_amdgcn_tensor_load_to_lds(g0, g1, z4, z4, z8, 0);
}

// ---------------------------------------------------------------------------
// Big input GEMM: Xcur[MB_X, H] = x_h[MB_X, 704] @ Wx[704, H]
// 256 threads = 8 waves (2Mx4N), wave tile 32x32 (2x2 WMMA).
// A tiles double-buffered in LDS via the Tensor Data Mover (wave 0 issues,
// TENSORcnt + workgroup barrier synchronize); W fragments from global (L2).
// ---------------------------------------------------------------------------
__global__ __launch_bounds__(256) void gemm_x_kernel(
    const _Float16* __restrict__ xh, const _Float16* __restrict__ Wx,
    float* __restrict__ xcur)
{
    __shared__ _Float16 lA[2][BM * LDA];
    const int tid  = threadIdx.x;
    const int wave = tid >> 5;
    const int lane = tid & 31;
    const int bm = blockIdx.x * BM;          // grid.x = MB_X/BM = 1000
    const int bn = blockIdx.y * BN;          // grid.y = H/BN = 4
    const int wm = (wave >> 2) * 32;         // 0 | 32
    const int wn = (wave & 3) * 32;          // 0..96
    const int arow = lane & 15;
    const int k0   = (lane < 16) ? 0 : 8;    // A-fragment K base (ISA layout)

    const unsigned ldsbase[2] = { (unsigned)(uintptr_t)&lA[0][0],
                                  (unsigned)(uintptr_t)&lA[1][0] };

    v8f acc[2][2] = {};

    // prologue: DMA tile kc=0 into buffer 0
    if (wave == 0) tdm_load_tile(&xh[(size_t)bm * KIN], ldsbase[0]);

    int ib = 0;
    for (int kc = 0; kc < KIN; kc += BK, ib ^= 1) {
        const bool more = (kc + BK) < KIN;
        if (wave == 0) {
            if (more) tdm_load_tile(&xh[(size_t)bm * KIN + kc + BK], ldsbase[ib ^ 1]);
            if (more) __builtin_amdgcn_s_wait_tensorcnt((short)1);  // current tile done
            else      __builtin_amdgcn_s_wait_tensorcnt((short)0);
        }
        __syncthreads();                         // buffer ib visible to all waves

        Frag16 bfr[2];
        #pragma unroll
        for (int nt = 0; nt < 2; ++nt)
            for (int s = 0; s < 16; ++s)
                bfr[nt].v[s] = Wx[(size_t)(kc + k0 + (s & 7) + ((s >> 3) << 4)) * H + bn + wn + nt * 16 + (lane & 15)];
        if (more && lane == 0)                   // near-scope prefetch of next W tile
            __builtin_prefetch(&Wx[(size_t)(kc + BK + wave * 4) * H + bn + wn], 0, 3);

        #pragma unroll
        for (int mt = 0; mt < 2; ++mt) {
            Frag16 afr;
            const _Float16* ap = &lA[ib][(wm + mt * 16 + arow) * LDA];
            afr.h[0] = *(const v8h*)(ap + k0);
            afr.h[1] = *(const v8h*)(ap + k0 + 16);
            #pragma unroll
            for (int nt = 0; nt < 2; ++nt)
                acc[mt][nt] = __builtin_amdgcn_wmma_f32_16x16x32_f16(
                    false, afr.v, false, bfr[nt].v, (short)0, acc[mt][nt], false, false);
        }
        __syncthreads();                         // all reads of buffer ib complete
    }

    // store C: element r -> (m0 + r + 8*(lane>=16), n0 + (lane&15))
    const int mrow0 = bm + wm + 8 * (lane >> 4);
    const int ncol0 = bn + wn + (lane & 15);
    #pragma unroll
    for (int mt = 0; mt < 2; ++mt)
        #pragma unroll
        for (int nt = 0; nt < 2; ++nt)
            #pragma unroll
            for (int r = 0; r < 8; ++r)
                xcur[(size_t)(mrow0 + mt * 16 + r) * H + ncol0 + nt * 16] = acc[mt][nt][r];
}

// ---------------------------------------------------------------------------
// One recurrent step: rec = z_in @ Wh ([256,512]@[512,512]) fused with the
// ALIF cell update, producing z_out (f16 spikes) + u/a states in place.
// grid (4,4): 64x128 block tiles, same wave layout as gemm_x.
// ---------------------------------------------------------------------------
__global__ __launch_bounds__(256) void rec_step_kernel(
    const _Float16* __restrict__ zin, const _Float16* __restrict__ Wh,
    const float* __restrict__ xcur_t,
    float* __restrict__ u, float* __restrict__ a,
    const float* __restrict__ alpha, const float* __restrict__ rho,
    _Float16* __restrict__ zout, float* __restrict__ spikes)
{
    const int tid  = threadIdx.x;
    const int wave = tid >> 5;
    const int lane = tid & 31;
    const int bm = blockIdx.x * BM;          // 4 blocks -> M=256
    const int bn = blockIdx.y * BN;          // 4 blocks -> N=512
    const int wm = (wave >> 2) * 32;
    const int wn = (wave & 3) * 32;
    const int arow = lane & 15;
    const int k0   = (lane < 16) ? 0 : 8;

    v8f acc[2][2] = {};

    for (int kc = 0; kc < H; kc += BK) {
        Frag16 bfr[2];
        #pragma unroll
        for (int nt = 0; nt < 2; ++nt)
            for (int s = 0; s < 16; ++s)
                bfr[nt].v[s] = Wh[(size_t)(kc + k0 + (s & 7) + ((s >> 3) << 4)) * H + bn + wn + nt * 16 + (lane & 15)];
        #pragma unroll
        for (int mt = 0; mt < 2; ++mt) {
            Frag16 afr;
            const _Float16* ap = &zin[(size_t)(bm + wm + mt * 16 + arow) * H + kc];
            afr.h[0] = *(const v8h*)(ap + k0);
            afr.h[1] = *(const v8h*)(ap + k0 + 16);
            #pragma unroll
            for (int nt = 0; nt < 2; ++nt)
                acc[mt][nt] = __builtin_amdgcn_wmma_f32_16x16x32_f16(
                    false, afr.v, false, bfr[nt].v, (short)0, acc[mt][nt], false, false);
        }
    }

    // fused ALIF elementwise update on accumulator fragments
    const int mrow0 = bm + wm + 8 * (lane >> 4);
    const int ncol0 = bn + wn + (lane & 15);
    float local_spikes = 0.0f;
    #pragma unroll
    for (int mt = 0; mt < 2; ++mt) {
        #pragma unroll
        for (int nt = 0; nt < 2; ++nt) {
            const int n = ncol0 + nt * 16;
            const float rho_n = rho[n];
            const float al_n  = alpha[n];
            #pragma unroll
            for (int r = 0; r < 8; ++r) {
                const int m = mrow0 + mt * 16 + r;
                const size_t idx = (size_t)m * H + n;
                const float zo    = (float)zin[idx];
                const float a_new = rho_n * a[idx] + (1.0f - rho_n) * zo;
                const float theta = 0.01f + 1.8f * a_new;
                const float cur   = xcur_t[idx] + acc[mt][nt][r];
                const float u_new = al_n * u[idx] + (1.0f - al_n) * cur - zo * theta;
                const float zn    = (u_new - theta > 0.0f) ? 1.0f : 0.0f;
                a[idx] = a_new;
                u[idx] = u_new;
                zout[idx] = (_Float16)zn;
                local_spikes += zn;
            }
        }
    }
    atomicAdd(spikes, local_spikes);
}

// ---------------------------------------------------------------------------
// Readout + NLL for one step. 1 block, 256 threads: thread b owns batch row b.
// ---------------------------------------------------------------------------
__global__ __launch_bounds__(256) void readout_kernel(
    const _Float16* __restrict__ z, const float* __restrict__ W_out,
    const int* __restrict__ y_t, const float* __restrict__ kappa,
    float* __restrict__ ou, float* __restrict__ outputs_t,
    float* __restrict__ loss)
{
    const int b = threadIdx.x;
    float acc[OUTSZ];
    #pragma unroll
    for (int j = 0; j < OUTSZ; ++j) acc[j] = 0.0f;

    const _Float16* zr = z + (size_t)b * H;
    for (int k8 = 0; k8 < H / 8; ++k8) {
        v8h zc = *(const v8h*)(zr + k8 * 8);
        #pragma unroll
        for (int e = 0; e < 8; ++e) {
            const float zv = (float)zc[e];
            const float* wr = W_out + (size_t)(k8 * 8 + e) * OUTSZ;
            #pragma unroll
            for (int j = 0; j < OUTSZ; ++j) acc[j] = fmaf(zv, wr[j], acc[j]);
        }
    }

    float lo[OUTSZ];
    float mx = -1e30f;
    #pragma unroll
    for (int j = 0; j < OUTSZ; ++j) {
        const float kp = kappa[j];
        const float v  = kp * ou[b * OUTSZ + j] + (1.0f - kp) * acc[j];
        lo[j] = v;
        ou[b * OUTSZ + j] = v;
        outputs_t[b * OUTSZ + j] = v;
        mx = fmaxf(mx, v);
    }
    float se = 0.0f;
    #pragma unroll
    for (int j = 0; j < OUTSZ; ++j) se += expf(lo[j] - mx);
    const float lse = mx + logf(se);
    const int lbl = y_t[b];
    const float lb = lse - lo[lbl];

    __shared__ float red[256];
    red[b] = lb;
    __syncthreads();
    for (int s = 128; s > 0; s >>= 1) {
        if (b < s) red[b] += red[b + s];
        __syncthreads();
    }
    if (b == 0) atomicAdd(loss, red[0] * (1.0f / (float)BATCH));
}

// ---------------------------------------------------------------------------
// host launcher
// ---------------------------------------------------------------------------
static inline size_t align256(size_t v) { return (v + 255) & ~(size_t)255; }

extern "C" void kernel_launch(void* const* d_in, const int* in_sizes, int n_in,
                              void* d_out, int out_size, void* d_ws, size_t ws_size,
                              hipStream_t stream) {
    (void)in_sizes; (void)n_in; (void)out_size; (void)ws_size;

    const float* x        = (const float*)d_in[0];   // [250,256,700]
    const int*   y        = (const int*)  d_in[1];   // [250,256]
    const float* W_hidden = (const float*)d_in[2];   // [1212,512]
    const float* tau_mem  = (const float*)d_in[3];   // [512]
    const float* tau_adp  = (const float*)d_in[4];   // [512]
    const float* W_out    = (const float*)d_in[5];   // [512,10]
    const float* tau_out  = (const float*)d_in[6];   // [10]
    float* out = (float*)d_out;                      // 640000 outputs + loss + spikes

    char* ws = (char*)d_ws;
    size_t off = 0;
    auto take = [&](size_t bytes) { size_t o = off; off = align256(off + bytes); return o; };
    _Float16* Wx    = (_Float16*)(ws + take((size_t)KIN * H * 2));
    _Float16* Wh    = (_Float16*)(ws + take((size_t)H * H * 2));
    float*    alpha = (float*)   (ws + take(H * 4));
    float*    rho   = (float*)   (ws + take(H * 4));
    float*    kappa = (float*)   (ws + take(OUTSZ * 4));
    _Float16* z0    = (_Float16*)(ws + take((size_t)BATCH * H * 2));
    _Float16* z1    = (_Float16*)(ws + take((size_t)BATCH * H * 2));
    float*    u     = (float*)   (ws + take((size_t)BATCH * H * 4));
    float*    a     = (float*)   (ws + take((size_t)BATCH * H * 4));
    float*    ou    = (float*)   (ws + take((size_t)BATCH * OUTSZ * 4));
    _Float16* xh    = (_Float16*)(ws + take((size_t)MB_X * KIN * 2));
    float*    xcur  = (float*)   (ws + take((size_t)MB_X * H * 4));

    float* loss_ptr   = out + (size_t)T_STEPS * BATCH * OUTSZ;       // 640000
    float* spikes_ptr = loss_ptr + 1;                                 // 640001

    init_kernel<<<256, 256, 0, stream>>>(tau_mem, tau_adp, tau_out,
                                         alpha, rho, kappa, z0, u, a, ou, loss_ptr);
    convert_w_kernel<<<512, 256, 0, stream>>>(W_hidden, Wx, Wh);
    convert_x_kernel<<<(MB_X * (KIN / 8)) / 256, 256, 0, stream>>>(x, xh);

    gemm_x_kernel<<<dim3(MB_X / BM, H / BN), 256, 0, stream>>>(xh, Wx, xcur);

    for (int t = 0; t < T_STEPS; ++t) {
        const _Float16* zin  = (t & 1) ? z1 : z0;
        _Float16*       zout = (t & 1) ? z0 : z1;
        rec_step_kernel<<<dim3(BATCH / BM, H / BN), 256, 0, stream>>>(
            zin, Wh, xcur + (size_t)t * BATCH * H, u, a, alpha, rho, zout, spikes_ptr);
        readout_kernel<<<1, 256, 0, stream>>>(
            zout, W_out, y + (size_t)t * BATCH, kappa, ou,
            out + (size_t)t * BATCH * OUTSZ, loss_ptr);
    }
}
